// MultiboxLoss_31430570672618
// MI455X (gfx1250) — compile-verified, matrix operand order checked by reading, not run on hardware
//
#include <hip/hip_runtime.h>
#include <stdint.h>

typedef __attribute__((ext_vector_type(16))) _Float16 v16h;
typedef __attribute__((ext_vector_type(8)))  float    v8f;

#define BATCH   16
#define NPRIOR  65536
#define NGT     16
#define PPB     1024                 // priors per block
#define TILES   (NPRIOR / PPB)       // 64
#define NBLK    (BATCH * TILES)      // 1024
#define NITER   (PPB / 16)           // 64 iterations of 16 priors per block

// ---- workspace byte offsets ----
#define WS_BEST   0u          // u64[BATCH*NGT] packed (iou_bits<<32)|prior_idx
#define WS_NUMPOS 2048u       // u32[16]
#define WS_SUMS   2112u       // f32[4]: 0=locSum 1=ceMatchSum 2=ceSelSum
#define WS_SEL0   2176u       // u32[16]
#define WS_REM1   2240u       // u32[16]
#define WS_SEL1   2304u       // u32[16]
#define WS_REM2   2368u       // u32[16]
#define WS_H0C    4096u       // u32[16*1024]
#define WS_H0S    69632u      // f32[16*1024]
#define WS_H1C    135168u     // u32[16*1024]
#define WS_H1S    200704u     // f32[16*1024]
#define WS_H2C    266240u     // u32[16*4096]
#define WS_H2S    528384u     // f32[16*4096]
#define WS_TOTAL  790528u

#define LN2_F     0.6931471805599453f
#define LOG2E_F   1.4426950408889634f

#if __has_builtin(__builtin_amdgcn_rcpf)
__device__ __forceinline__ float fast_rcp(float x) { return __builtin_amdgcn_rcpf(x); }
#else
__device__ __forceinline__ float fast_rcp(float x) { return 1.0f / x; }
#endif

#if __has_builtin(__builtin_amdgcn_logf)
__device__ __forceinline__ float fast_log2(float x) { return __builtin_amdgcn_logf(x); }
#else
__device__ __forceinline__ float fast_log2(float x) { return __logf(x) * LOG2E_F; }
#endif

#if __has_builtin(__builtin_amdgcn_exp2f)
__device__ __forceinline__ float fast_exp2(float x) { return __builtin_amdgcn_exp2f(x); }
#else
__device__ __forceinline__ float fast_exp2(float x) { return exp2f(x); }
#endif

__device__ __forceinline__ uint32_t sortKey(float f) {
    uint32_t u = __float_as_uint(f);
    return (u & 0x80000000u) ? ~u : (u | 0x80000000u);
}

// log1p(exp(-|x|)) = ln2 * log2(1 + 2^(-|x|*log2e)); arg in (1,2], never denormal
__device__ __forceinline__ float softplus_neg_abs(float x) {
    float z = fast_exp2(-fabsf(x) * LOG2E_F);
    return fast_log2(1.0f + z) * LN2_F;
}

__device__ __forceinline__ float iou_fn(float4 pr, float pa,
                                        float gx0, float gy0, float gx1, float gy1,
                                        float ga) {
    float ltx = fmaxf(pr.x, gx0), lty = fmaxf(pr.y, gy0);
    float rbx = fminf(pr.z, gx1), rby = fminf(pr.w, gy1);
    float w = fminf(fmaxf(rbx - ltx, 0.0f), 100.0f);
    float h = fminf(fmaxf(rby - lty, 0.0f), 100.0f);
    float inter = w * h;
    return inter * fast_rcp(pa + ga - inter);   // union > 0 always
}

// -------- zero workspace --------
__global__ void k_init(uint32_t* ws) {
    uint32_t n = WS_TOTAL / 4u;
    for (uint32_t i = blockIdx.x * blockDim.x + threadIdx.x; i < n;
         i += gridDim.x * blockDim.x)
        ws[i] = 0u;
}

// -------- pass A: per-(b,g) argmax of IoU over priors --------
__global__ void k_best(const float4* __restrict__ prior,
                       const float4* __restrict__ gt,
                       char* __restrict__ ws) {
    int b = blockIdx.x >> 6;           // TILES == 64
    int tile = blockIdx.x & 63;
    int t = threadIdx.x;
    int g = t & 15;
    int slot = t >> 4;
    int pBase = tile * PPB;
    const float4* priorB = prior + (size_t)b * NPRIOR;

    float4 gb = gt[b * NGT + g];
    float ga = (gb.z - gb.x) * (gb.w - gb.y);

    unsigned long long bestLoc = 0ull;
    for (int it = 0; it < NITER; ++it) {
        int p = pBase + it * 16 + slot;
        float4 pr = priorB[p];
        float pa = (pr.z - pr.x) * (pr.w - pr.y);
        float iou = iou_fn(pr, pa, gb.x, gb.y, gb.z, gb.w, ga);
        unsigned long long packed =
            ((unsigned long long)__float_as_uint(iou) << 32) | (uint32_t)p;
        bestLoc = (packed > bestLoc) ? packed : bestLoc;
    }

    __shared__ unsigned long long sBest[16];
    if (t < 16) sBest[t] = 0ull;
    __syncthreads();
    atomicMax(&sBest[g], bestLoc);
    __syncthreads();
    if (t < 16) {
        unsigned long long* best = (unsigned long long*)(ws + WS_BEST);
        atomicMax(&best[b * NGT + t], sBest[t]);
    }
}

// -------- pass B: match/huber/ce + WMMA per-g counts + level-0 histogram --------
__global__ void k_main(const float4* __restrict__ prior,
                       const float4* __restrict__ reg,
                       const float*  __restrict__ conf,
                       const float4* __restrict__ gt,
                       char* __restrict__ ws) {
    int b = blockIdx.x >> 6;
    int tile = blockIdx.x & 63;
    int t = threadIdx.x;
    int g = t & 15;
    int slot = t >> 4;
    int lane = t & 31;
    int pBase = tile * PPB;
    const float4* priorB = prior + (size_t)b * NPRIOR;
    const float4* regB   = reg   + (size_t)b * NPRIOR;
    const float*  confB  = conf  + (size_t)b * NPRIOR;

    float4 gb = gt[b * NGT + g];
    float gw = gb.z - gb.x, gh = gb.w - gb.y;
    float ga = gw * gh;
    float gcx = gb.x + 0.5f * gw, gcy = gb.y + 0.5f * gh;
    float l2gw = fast_log2(gw), l2gh = fast_log2(gh);   // y2 = (l2gw - log2(pw)) * ln2

    unsigned long long bp = ((const unsigned long long*)(ws + WS_BEST))[b * NGT + g];
    uint32_t bestIdx = (uint32_t)bp;
    bool bestPos = (uint32_t)(bp >> 32) != 0u;

    uint32_t* h0c = (uint32_t*)(ws + WS_H0C) + g * 1024;
    float*    h0s = (float*)   (ws + WS_H0S) + g * 1024;

    float locAcc = 0.0f, ceMAcc = 0.0f;
    v8f cnt = {0.f, 0.f, 0.f, 0.f, 0.f, 0.f, 0.f, 0.f};
    v16h ones;
#pragma unroll
    for (int i = 0; i < 16; ++i) ones[i] = (_Float16)1.0f;

    for (int outer = 0; outer < NITER / 16; ++outer) {        // 4 outers
        // prefetch next chunk (gfx1250 global_prefetch_b8)
        if (outer < 3) {
            int pn = pBase + (outer + 1) * 256 + slot * 16;
            __builtin_prefetch((const void*)(priorB + pn), 0, 1);
            __builtin_prefetch((const void*)(regB + pn), 0, 1);
        }
        v16h bm;
#pragma unroll
        for (int inner = 0; inner < 16; ++inner) {
            int it = outer * 16 + inner;
            int p = pBase + it * 16 + slot;
            float4 pr = priorB[p];
            float4 rg = regB[p];
            float x = confB[p];

            float pw = pr.z - pr.x, ph = pr.w - pr.y;
            float pa = pw * ph;
            float pcx = pr.x + 0.5f * pw, pcy = pr.y + 0.5f * ph;

            float iou = iou_fn(pr, pa, gb.x, gb.y, gb.z, gb.w, ga);
            bool match = (iou >= 0.5f) || (bestPos && ((uint32_t)p == bestIdx));

            // huber(y - regression), mean over 4 coords
            float y0 = (gcx - pcx) * fast_rcp(pw);
            float y1 = (gcy - pcy) * fast_rcp(ph);
            float y2 = (l2gw - fast_log2(pw)) * LN2_F;
            float y3 = (l2gh - fast_log2(ph)) * LN2_F;
            float hsum = 0.0f;
            {
                float e = fabsf(y0 - rg.x); float q = fminf(e, 1.0f);
                hsum += 0.5f * q * q + (e - q);
                e = fabsf(y1 - rg.y); q = fminf(e, 1.0f);
                hsum += 0.5f * q * q + (e - q);
                e = fabsf(y2 - rg.z); q = fminf(e, 1.0f);
                hsum += 0.5f * q * q + (e - q);
                e = fabsf(y3 - rg.w); q = fminf(e, 1.0f);
                hsum += 0.5f * q * q + (e - q);
            }
            float huber = 0.25f * hsum;

            float base = fmaxf(x, 0.0f) + softplus_neg_abs(x);
            float ce = match ? (base - x) : base;
            float masked = match ? -ce : ce;

            if (match) { locAcc += huber; ceMAcc += ce; }
            bm[inner] = match ? (_Float16)1.0f : (_Float16)0.0f;

            uint32_t key = sortKey(masked);
            uint32_t bin = key >> 22;
            atomicAdd(&h0c[bin], 1u);
            if (!match) atomicAdd(&h0s[bin], masked);
        }
        // per-wave per-g match count: D = ones(16x32) x B(32x16) + C
        cnt = __builtin_amdgcn_wmma_f32_16x16x32_f16(
            false, ones, false, bm, (short)0, cnt, false, false);
    }

    // cnt[0] in lanes 0..15 holds the column (=g) sums for this wave's 128 priors
    if (lane < 16) {
        uint32_t* np = (uint32_t*)(ws + WS_NUMPOS);
        atomicAdd(&np[lane], (uint32_t)(cnt[0] + 0.5f));
    }
    // scalar sums: full-wave tree reduction, then one global atomic per wave
    for (int off = 16; off > 0; off >>= 1) {
        locAcc += __shfl_xor(locAcc, off, 32);
        ceMAcc += __shfl_xor(ceMAcc, off, 32);
    }
    if (lane == 0) {
        float* sums = (float*)(ws + WS_SUMS);
        atomicAdd(&sums[0], locAcc);
        atomicAdd(&sums[1], ceMAcc);
    }
}

// -------- radix-select scan over a histogram level --------
__global__ void k_select(char* __restrict__ ws, int level) {
    int g = threadIdx.x;
    if (g >= 16) return;
    uint32_t k;
    const uint32_t* hc;
    const float* hs;
    if (level == 0) {
        uint32_t np = ((const uint32_t*)(ws + WS_NUMPOS))[g];
        long long kk = 3ll * (long long)np;
        if (kk < 1) kk = 1;
        long long tot = (long long)BATCH * NPRIOR;
        if (kk > tot) kk = tot;
        k = (uint32_t)kk;
        hc = (const uint32_t*)(ws + WS_H0C) + g * 1024;
        hs = (const float*)(ws + WS_H0S) + g * 1024;
    } else {
        k = ((const uint32_t*)(ws + WS_REM1))[g];
        hc = (const uint32_t*)(ws + WS_H1C) + g * 1024;
        hs = (const float*)(ws + WS_H1S) + g * 1024;
    }
    uint32_t cum = 0, bin = 0, rem = 1;
    float sAbove = 0.0f;
    for (int i = 1023; i >= 0; --i) {
        uint32_t c = hc[i];
        if (cum + c >= k) { bin = (uint32_t)i; rem = k - cum; break; }
        cum += c;
        sAbove += hs[i];
    }
    if (level == 0) {
        ((uint32_t*)(ws + WS_SEL0))[g] = bin;
        ((uint32_t*)(ws + WS_REM1))[g] = rem;
    } else {
        ((uint32_t*)(ws + WS_SEL1))[g] = bin;
        ((uint32_t*)(ws + WS_REM2))[g] = rem;
    }
    atomicAdd(&((float*)(ws + WS_SUMS))[2], sAbove);
}

// -------- refine passes: histogram restricted to the selected bin path --------
__global__ void k_lvl(const float4* __restrict__ prior,
                      const float*  __restrict__ conf,
                      const float4* __restrict__ gt,
                      char* __restrict__ ws, int level) {
    int b = blockIdx.x >> 6;
    int tile = blockIdx.x & 63;
    int t = threadIdx.x;
    int g = t & 15;
    int slot = t >> 4;
    int pBase = tile * PPB;
    const float4* priorB = prior + (size_t)b * NPRIOR;
    const float*  confB  = conf  + (size_t)b * NPRIOR;

    float4 gb = gt[b * NGT + g];
    float ga = (gb.z - gb.x) * (gb.w - gb.y);

    unsigned long long bp = ((const unsigned long long*)(ws + WS_BEST))[b * NGT + g];
    uint32_t bestIdx = (uint32_t)bp;
    bool bestPos = (uint32_t)(bp >> 32) != 0u;

    uint32_t sel0 = ((const uint32_t*)(ws + WS_SEL0))[g];
    bool lvl2 = (level == 2);
    uint32_t want = lvl2 ? ((sel0 << 10) | ((const uint32_t*)(ws + WS_SEL1))[g]) : sel0;
    uint32_t* hc = lvl2 ? (uint32_t*)(ws + WS_H2C) + g * 4096
                        : (uint32_t*)(ws + WS_H1C) + g * 1024;
    float* hsum = lvl2 ? (float*)(ws + WS_H2S) + g * 4096
                       : (float*)(ws + WS_H1S) + g * 1024;

    for (int it = 0; it < NITER; ++it) {
        int p = pBase + it * 16 + slot;
        float4 pr = priorB[p];
        float x = confB[p];
        float pa = (pr.z - pr.x) * (pr.w - pr.y);
        float iou = iou_fn(pr, pa, gb.x, gb.y, gb.z, gb.w, ga);
        bool match = (iou >= 0.5f) || (bestPos && ((uint32_t)p == bestIdx));
        float base = fmaxf(x, 0.0f) + softplus_neg_abs(x);
        float masked = match ? -(base - x) : base;
        uint32_t key = sortKey(masked);
        uint32_t hi = lvl2 ? (key >> 12) : (key >> 22);
        if (hi == want) {
            uint32_t bin = lvl2 ? (key & 4095u) : ((key >> 12) & 1023u);
            atomicAdd(&hc[bin], 1u);
            if (!match) atomicAdd(&hsum[bin], masked);
        }
    }
}

// -------- final: scan level-2, assemble scalar loss --------
__global__ void k_final(char* __restrict__ ws, float* __restrict__ out) {
    __shared__ float selAdd[16];
    int t = threadIdx.x;
    if (t < 16) {
        const uint32_t* h2c = (const uint32_t*)(ws + WS_H2C) + t * 4096;
        const float*    h2s = (const float*)(ws + WS_H2S) + t * 4096;
        uint32_t k = ((const uint32_t*)(ws + WS_REM2))[t];
        uint32_t cum = 0;
        float s = 0.0f, add = 0.0f;
        for (int i = 4095; i >= 0; --i) {
            uint32_t c = h2c[i];
            if (cum + c >= k) { add = s + h2s[i]; break; }  // all keys in bin == thr
            cum += c;
            s += h2s[i];
        }
        selAdd[t] = add;
    }
    __syncthreads();
    if (t == 0) {
        const float* sums = (const float*)(ws + WS_SUMS);
        float ceSel = sums[2];
        for (int i = 0; i < 16; ++i) ceSel += selAdd[i];
        const uint32_t* np = (const uint32_t*)(ws + WS_NUMPOS);
        float tp = 0.0f;
        for (int i = 0; i < 16; ++i) tp += (float)np[i];
        out[0] = (sums[0] + sums[1] + ceSel) / tp;
    }
}

extern "C" void kernel_launch(void* const* d_in, const int* in_sizes, int n_in,
                              void* d_out, int out_size, void* d_ws, size_t ws_size,
                              hipStream_t stream) {
    (void)in_sizes; (void)n_in; (void)out_size; (void)ws_size;
    const float*  conf  = (const float*) d_in[0];
    const float4* reg   = (const float4*)d_in[1];
    const float4* prior = (const float4*)d_in[2];
    const float4* gt    = (const float4*)d_in[3];
    char* ws = (char*)d_ws;
    float* out = (float*)d_out;

    k_init<<<512, 256, 0, stream>>>((uint32_t*)ws);
    k_best<<<NBLK, 256, 0, stream>>>(prior, gt, ws);
    k_main<<<NBLK, 256, 0, stream>>>(prior, reg, conf, gt, ws);
    k_select<<<1, 16, 0, stream>>>(ws, 0);
    k_lvl<<<NBLK, 256, 0, stream>>>(prior, conf, gt, ws, 1);
    k_select<<<1, 16, 0, stream>>>(ws, 1);
    k_lvl<<<NBLK, 256, 0, stream>>>(prior, conf, gt, ws, 2);
    k_final<<<1, 64, 0, stream>>>(ws, out);
}
